// PointConvD_70437463654601
// MI455X (gfx1250) — compile-verified, matrix operand order checked by reading, not run on hardware
//
#include <hip/hip_runtime.h>

#define B_    8
#define N_    8192
#define D_    64
#define S_    1024
#define K_    16
#define WN_   16
#define OUT_  128
#define CIN_  67
#define KFLAT 1072          // CIN_ * WN_
#define KPAD  1088          // padded to multiple of 32 for 16x16x32 WMMA

typedef __attribute__((ext_vector_type(16))) _Float16 v16h;
typedef __attribute__((ext_vector_type(8)))  _Float16 v8h;
typedef __attribute__((ext_vector_type(8)))  float    v8f;

// ---------------------------------------------------------------------------
// Kernel 1: Furthest point sampling. One 1024-thread block per batch.
// Each thread owns 8 points in registers; per-iteration two-level wave32
// argmax reduction with first-index tie-break (matches jnp.argmax).
// Writes: fps indices (ws int + d_out float region), new_xyz ([B,3,S] at d_out).
// ---------------------------------------------------------------------------
__global__ __launch_bounds__(1024) void fps_kernel(const float* __restrict__ xyz,
                                                   int* __restrict__ fps_ws,
                                                   float* __restrict__ out) {
    const int b   = blockIdx.x;
    const int tid = threadIdx.x;
    const float* xb = xyz + (size_t)b * 3 * N_;

    __shared__ float s_c[3];
    __shared__ int   s_far;
    __shared__ float s_rv[32];
    __shared__ int   s_ri[32];

    float px[8], py[8], pz[8], dist[8];
#pragma unroll
    for (int j = 0; j < 8; ++j) {
        int n = tid + j * 1024;
        px[j] = xb[n];
        py[j] = xb[N_ + n];
        pz[j] = xb[2 * N_ + n];
        dist[j] = 1e10f;
    }

    int curF = 0;
    for (int i = 0; i < S_; ++i) {
        if (tid == 0) {
            float cx = xb[curF], cy = xb[N_ + curF], cz = xb[2 * N_ + curF];
            s_c[0] = cx; s_c[1] = cy; s_c[2] = cz;
            out[(size_t)b * 3 * S_ + 0 * S_ + i] = cx;
            out[(size_t)b * 3 * S_ + 1 * S_ + i] = cy;
            out[(size_t)b * 3 * S_ + 2 * S_ + i] = cz;
            fps_ws[b * S_ + i] = curF;
            out[(size_t)B_ * 3 * S_ + (size_t)B_ * OUT_ * S_ + b * S_ + i] = (float)curF;
        }
        __syncthreads();
        float cx = s_c[0], cy = s_c[1], cz = s_c[2];

        float bv = -1.0f; int bi = 0;
#pragma unroll
        for (int j = 0; j < 8; ++j) {
            float dx = px[j] - cx, dy = py[j] - cy, dz = pz[j] - cz;
            float d = dx * dx + dy * dy + dz * dz;
            dist[j] = fminf(dist[j], d);
            int n = tid + j * 1024;
            if (dist[j] > bv) { bv = dist[j]; bi = n; }
        }
        // wave32 butterfly argmax (prefer lower index on ties)
#pragma unroll
        for (int off = 16; off > 0; off >>= 1) {
            float ov = __shfl_xor(bv, off, 32);
            int   oi = __shfl_xor(bi, off, 32);
            if (ov > bv || (ov == bv && oi < bi)) { bv = ov; bi = oi; }
        }
        if ((tid & 31) == 0) { s_rv[tid >> 5] = bv; s_ri[tid >> 5] = bi; }
        __syncthreads();
        if (tid < 32) {
            bv = s_rv[tid]; bi = s_ri[tid];
#pragma unroll
            for (int off = 16; off > 0; off >>= 1) {
                float ov = __shfl_xor(bv, off, 32);
                int   oi = __shfl_xor(bi, off, 32);
                if (ov > bv || (ov == bv && oi < bi)) { bv = ov; bi = oi; }
            }
            if (tid == 0) s_far = bi;
        }
        __syncthreads();
        curF = s_far;
    }
}

// ---------------------------------------------------------------------------
// Kernel 2: KNN (16 nearest of 8192) per sample. One thread per sample,
// xyz staged through LDS in 512-point tiles, branchless unrolled insertion
// into a sorted 16-entry register list (ascending dist, stable -> matches
// top_k tie-breaking).
// ---------------------------------------------------------------------------
__global__ __launch_bounds__(256) void knn_kernel(const float* __restrict__ xyz,
                                                  const float* __restrict__ newxyz,
                                                  int* __restrict__ knn_ws) {
    const int b = blockIdx.x >> 2;
    const int s = ((blockIdx.x & 3) << 8) + threadIdx.x;
    const float* xb = xyz + (size_t)b * 3 * N_;

    float cx = newxyz[(size_t)b * 3 * S_ + s];
    float cy = newxyz[(size_t)b * 3 * S_ + S_ + s];
    float cz = newxyz[(size_t)b * 3 * S_ + 2 * S_ + s];

    __shared__ float tx[512], ty[512], tz[512];

    float d16[16]; int id16[16];
#pragma unroll
    for (int j = 0; j < 16; ++j) { d16[j] = 3.0e38f; id16[j] = 0; }

    for (int t0 = 0; t0 < N_; t0 += 512) {
        __syncthreads();
        for (int r = threadIdx.x; r < 512; r += 256) {
            tx[r] = xb[t0 + r];
            ty[r] = xb[N_ + t0 + r];
            tz[r] = xb[2 * N_ + t0 + r];
        }
        __syncthreads();
        for (int r = 0; r < 512; ++r) {
            float dx = tx[r] - cx, dy = ty[r] - cy, dz = tz[r] - cz;
            float dd = dx * dx + dy * dy + dz * dz;
            if (dd < d16[15]) {
                int n = t0 + r;
                int p = 15;
#pragma unroll
                for (int t = 15; t >= 1; --t) {
                    if (dd < d16[t - 1]) {
                        d16[t] = d16[t - 1]; id16[t] = id16[t - 1]; p = t - 1;
                    }
                }
                d16[p] = dd; id16[p] = n;
            }
        }
    }
    int base = (b * S_ + s) * 16;
#pragma unroll
    for (int j = 0; j < 16; ++j) knn_ws[base + j] = id16[j];
}

// ---------------------------------------------------------------------------
// Kernel 3: gather + WeightNet MLP + aggregation. One wave per sample,
// 8 samples per 256-thread block. feat[16x67] and wts[16x16] staged in LDS,
// then agg[c,w] = sum_k feat[k,c]*wts[k,w] written as f16 WMMA A-rows
// (K padded 1072 -> 1088 with zeros).
// ---------------------------------------------------------------------------
__global__ __launch_bounds__(256) void gwa_kernel(
    const float* __restrict__ xyz, const float* __restrict__ points,
    const float* __restrict__ newxyz, const int* __restrict__ knn_ws,
    const float* __restrict__ w1, const float* __restrict__ b1,
    const float* __restrict__ w2, const float* __restrict__ b2,
    const float* __restrict__ w3, const float* __restrict__ b3,
    _Float16* __restrict__ Af16) {
    __shared__ float sfeat[8][16][68];
    __shared__ float swts[8][16][16];
    __shared__ float s_w1[24], s_b1[8], s_w2[64], s_b2[8], s_w3[128], s_b3[16];

    const int tid = threadIdx.x;
    if (tid < 24)  s_w1[tid] = w1[tid];
    if (tid < 8)   { s_b1[tid] = b1[tid]; s_b2[tid] = b2[tid]; }
    if (tid < 64)  s_w2[tid] = w2[tid];
    if (tid < 128) s_w3[tid] = w3[tid];
    if (tid < 16)  s_b3[tid] = b3[tid];

    const int samp = tid >> 5;
    const int lane = tid & 31;
    const int k    = lane & 15;
    const int half = lane >> 4;
    const int m    = blockIdx.x * 8 + samp;   // global sample id, row of A
    const int b    = m >> 10;
    const int s    = m & 1023;

    const int nidx = knn_ws[(m << 4) + k];
    const float* xb = xyz + (size_t)b * 3 * N_;
    const float* pb = points + (size_t)b * D_ * N_;

    // gather 64 point features: half 0 -> d 0..31, half 1 -> d 32..63
#pragma unroll 8
    for (int dd = 0; dd < 32; ++dd) {
        int d = half * 32 + dd;
        sfeat[samp][k][3 + d] = pb[(size_t)d * N_ + nidx];
    }

    float dx = 0.f, dy = 0.f, dz = 0.f;
    if (half == 0) {
        float cx = newxyz[(size_t)b * 3 * S_ + s];
        float cy = newxyz[(size_t)b * 3 * S_ + S_ + s];
        float cz = newxyz[(size_t)b * 3 * S_ + 2 * S_ + s];
        dx = xb[nidx] - cx;
        dy = xb[N_ + nidx] - cy;
        dz = xb[2 * N_ + nidx] - cz;
        sfeat[samp][k][0] = dx; sfeat[samp][k][1] = dy; sfeat[samp][k][2] = dz;
    }
    __syncthreads();   // weights + feat visible

    if (half == 0) {
        float h1[8];
#pragma unroll
        for (int j = 0; j < 8; ++j)
            h1[j] = fmaxf(0.f, dx * s_w1[j] + dy * s_w1[8 + j] + dz * s_w1[16 + j] + s_b1[j]);
        float h2[8];
#pragma unroll
        for (int j = 0; j < 8; ++j) {
            float a = s_b2[j];
#pragma unroll
            for (int i2 = 0; i2 < 8; ++i2) a += h1[i2] * s_w2[i2 * 8 + j];
            h2[j] = fmaxf(0.f, a);
        }
#pragma unroll
        for (int w = 0; w < 16; ++w) {
            float a = s_b3[w];
#pragma unroll
            for (int i2 = 0; i2 < 8; ++i2) a += h2[i2] * s_w3[i2 * 16 + w];
            swts[samp][k][w] = fmaxf(0.f, a);
        }
    }
    __syncthreads();

    // agg[c*16+w] = sum_k feat[k][c] * wts[k][w]   -> f16 row of A
    _Float16* arow = Af16 + (size_t)m * KPAD;
    for (int cw = lane; cw < KFLAT; cw += 32) {
        int c = cw >> 4, w = cw & 15;
        float a = 0.f;
#pragma unroll
        for (int kk = 0; kk < 16; ++kk) a += sfeat[samp][kk][c] * swts[samp][kk][w];
        arow[cw] = (_Float16)a;
    }
    if (lane < 16) arow[KFLAT + lane] = (_Float16)0.f;
}

// ---------------------------------------------------------------------------
// Kernel 4: lin_w [1072,128] f32 -> WT [128,1088] f16 (transposed, zero pad)
// ---------------------------------------------------------------------------
__global__ void wt_kernel(const float* __restrict__ lin_w, _Float16* __restrict__ WT) {
    int tid = blockIdx.x * blockDim.x + threadIdx.x;
    if (tid >= OUT_ * KPAD) return;
    int o = tid / KPAD, kk = tid % KPAD;
    float v = (kk < KFLAT) ? lin_w[(size_t)kk * OUT_ + o] : 0.f;
    WT[tid] = (_Float16)v;
}

// ---------------------------------------------------------------------------
// Kernel 5: WMMA GEMM  out[8192,128] = A[8192,1088]f16 x W[1088,128]f16
// + bias + LeakyReLU(0.1), stored transposed as [B, OUT, S].
//
// Block = 8 waves sharing one 16-col B tile (16 x 1088 f16 = 34 KB), staged
// into LDS once with GLOBAL_LOAD_ASYNC_TO_LDS_B128 (ASYNCcnt path), then each
// wave runs a 16x16 tile over 34 k-steps of v_wmma_f32_16x16x32_f16 with
// A streamed from global (+prefetch) and B fragments via ds_load.
// ---------------------------------------------------------------------------
#define TILE_B_HALFS (16 * KPAD)   // 17408 halves = 34816 bytes

__global__ __launch_bounds__(256) void gemm_kernel(const _Float16* __restrict__ A,
                                                   const _Float16* __restrict__ WT,
                                                   const float* __restrict__ bias,
                                                   float* __restrict__ out) {
    __shared__ _Float16 sB[TILE_B_HALFS];

    const int ntile  = blockIdx.x & 7;      // 8 col tiles
    const int mgroup = blockIdx.x >> 3;     // 64 groups of 8 row tiles
    const int tid    = threadIdx.x;
    const int wave   = tid >> 5;
    const int lane   = tid & 31;
    const int l16    = lane & 15;
    const int hi     = lane >> 4;

    // ---- async-stage the shared B tile: 2176 x 16B transfers across block ----
    {
        const _Float16* src = WT + (size_t)ntile * 16 * KPAD;
        for (int i = tid; i < TILE_B_HALFS / 8; i += 256) {
            unsigned           ldsa = (unsigned)(size_t)(&sB[i * 8]);
            unsigned long long ga   = (unsigned long long)(const void*)(src + i * 8);
            asm volatile("global_load_async_to_lds_b128 %0, %1, off"
                         :: "v"(ldsa), "v"(ga) : "memory");
        }
        asm volatile("s_wait_asynccnt 0x0" ::: "memory");
    }
    __syncthreads();

    const int mtile = mgroup * 8 + wave;    // 0..511
    const _Float16* arow = A + (size_t)(mtile * 16 + l16) * KPAD + hi * 8;
    const _Float16* brow = &sB[(size_t)l16 * KPAD + hi * 8];

    v8f acc = {};
#pragma unroll 2
    for (int kk = 0; kk < KPAD / 32; ++kk) {
        v8h a0 = *(const v8h*)(arow + kk * 32);
        v8h a1 = *(const v8h*)(arow + kk * 32 + 16);
        v8h b0 = *(const v8h*)(brow + kk * 32);
        v8h b1 = *(const v8h*)(brow + kk * 32 + 16);
        v16h av = __builtin_shufflevector(a0, a1, 0,1,2,3,4,5,6,7,8,9,10,11,12,13,14,15);
        v16h bv = __builtin_shufflevector(b0, b1, 0,1,2,3,4,5,6,7,8,9,10,11,12,13,14,15);
        acc = __builtin_amdgcn_wmma_f32_16x16x32_f16(
            /*neg_a=*/false, av, /*neg_b=*/false, bv,
            /*c_mod=*/(short)0, acc, /*reuse_a=*/false, /*reuse_b=*/false);
        __builtin_prefetch(arow + kk * 32 + 64, 0, 1);
    }

    const int   n   = ntile * 16 + l16;
    const float bvs = bias[n];
    const int   mb  = mtile * 16 + hi * 8;
#pragma unroll
    for (int r = 0; r < 8; ++r) {
        int m = mb + r;
        float v = acc[r] + bvs;
        v = (v > 0.f) ? v : 0.1f * v;
        int bb = m >> 10, ss = m & 1023;
        out[(size_t)bb * OUT_ * S_ + (size_t)n * S_ + ss] = v;
    }
}

// ---------------------------------------------------------------------------
extern "C" void kernel_launch(void* const* d_in, const int* in_sizes, int n_in,
                              void* d_out, int out_size, void* d_ws, size_t ws_size,
                              hipStream_t stream) {
    (void)in_sizes; (void)n_in; (void)out_size; (void)ws_size;

    const float* xyz    = (const float*)d_in[0];
    const float* points = (const float*)d_in[1];
    const float* w1     = (const float*)d_in[2];
    const float* b1     = (const float*)d_in[3];
    const float* w2     = (const float*)d_in[4];
    const float* b2     = (const float*)d_in[5];
    const float* w3     = (const float*)d_in[6];
    const float* b3     = (const float*)d_in[7];
    const float* lin_w  = (const float*)d_in[8];
    const float* lin_b  = (const float*)d_in[9];

    float* out = (float*)d_out;

    char* ws = (char*)d_ws;
    int*      fps_ws = (int*)ws;                                   //  32 KB
    int*      knn_ws = (int*)(ws + (32 << 10));                    // 512 KB
    _Float16* Af16   = (_Float16*)(ws + (32 << 10) + (512 << 10)); //  17 MB
    _Float16* WT     = (_Float16*)(ws + (32 << 10) + (512 << 10)
                                   + (size_t)B_ * S_ * KPAD * sizeof(_Float16));

    float* newxyz  = out;                          // [B,3,S]
    float* outFeat = out + (size_t)B_ * 3 * S_;    // [B,OUT,S]

    fps_kernel <<<B_,   1024, 0, stream>>>(xyz, fps_ws, out);
    knn_kernel <<<B_*4,  256, 0, stream>>>(xyz, newxyz, knn_ws);
    wt_kernel  <<<(OUT_ * KPAD + 255) / 256, 256, 0, stream>>>(lin_w, WT);
    gwa_kernel <<<(B_ * S_) / 8, 256, 0, stream>>>(xyz, points, newxyz, knn_ws,
                                                   w1, b1, w2, b2, w3, b3, Af16);
    gemm_kernel<<<(B_ * S_ / 16) * (OUT_ / 16) / 8, 256, 0, stream>>>(Af16, WT, lin_b, outFeat);
}
